// Model_68186900792054
// MI455X (gfx1250) — matrix operand order repaired and checked by reading, not compiled, not run on hardware
//
#include <hip/hip_runtime.h>
#include <stdint.h>

// Chunk-local reverse inclusive cumsum with fused scale, as C = M @ X with
// M[t,s] = 0.5*[s>=t] on gfx1250 WMMA (v_wmma_f32_16x16x32_f16), f16 hi/lo
// split of X for ~fp32 accuracy. Memory-bound: 128 MiB @ 23.3 TB/s ~= 5.5us.
// Staging uses CDNA5 async direct-to-LDS (ASYNCcnt); each wave owns one
// 16-column head block so B tiles are read once and reused across row-tiles.

typedef __attribute__((ext_vector_type(16))) _Float16 v16h;
typedef __attribute__((ext_vector_type(8)))  float    v8f;
typedef __attribute__((ext_vector_type(4)))  int      v4i;

#define BT 64
#define NH 64

#if defined(__has_builtin)
#  if __has_builtin(__builtin_amdgcn_global_load_async_to_lds_b128)
#    define USE_ASYNC_LDS 1
#  endif
#endif
#ifndef USE_ASYNC_LDS
#  define USE_ASYNC_LDS 0
#endif

#if USE_ASYNC_LDS
#  define LDS_STRIDE 64   // rows 16B-aligned for async b128 direct-to-LDS
#else
#  define LDS_STRIDE 65   // +1 pad: conflict-free column reads
#endif

#define AS3 __attribute__((address_space(3)))

__global__ __launch_bounds__(128, 1)
void chunk_rev_cumsum_wmma(const float* __restrict__ g, float* __restrict__ out) {
    __shared__ float xs[BT][LDS_STRIDE];

    const unsigned chunk = blockIdx.x;                 // flat over B*NT
    const size_t   base  = (size_t)chunk * (BT * NH);  // chunks are contiguous
    const int      tid   = threadIdx.x;

#if USE_ASYNC_LDS
    {
        // CDNA5 async direct-to-LDS staging (ASYNCcnt), no VGPR round-trip.
        const float* gp = g + base;
        #pragma unroll
        for (int it = 0; it < 8; ++it) {
            int f = tid + it * 128;                    // float4 index 0..1023
            v4i* gsrc = (v4i*)(gp + f * 4);
            AS3 v4i* ldst = (AS3 v4i*)(unsigned)(uintptr_t)(&xs[0][0] + f * 4);
            __builtin_amdgcn_global_load_async_to_lds_b128(gsrc, ldst, 0, 0);
        }
#  if defined(__has_builtin) && __has_builtin(__builtin_amdgcn_s_wait_asynccnt)
        __builtin_amdgcn_s_wait_asynccnt(0);
#  else
        asm volatile("s_wait_asynccnt 0" ::: "memory");
#  endif
    }
#else
    {
        // Fallback staging: coalesced global b128 loads + LDS stores.
        const float4* gin = (const float4*)(g + base);
        #pragma unroll
        for (int it = 0; it < 8; ++it) {
            int f = tid + it * 128;
            float4 v = gin[f];
            int row = f >> 4;
            int col = (f & 15) << 2;
            xs[row][col + 0] = v.x;
            xs[row][col + 1] = v.y;
            xs[row][col + 2] = v.z;
            xs[row][col + 3] = v.w;
        }
    }
#endif
    __syncthreads();

    const int lane = tid & 31;
    const int jw   = tid >> 5;        // this wave's head block: cols [16jw,16jw+16)
    const int ln15 = lane & 15;
    const int grp  = lane >> 4;

    v8f acc[4] = {v8f{}, v8f{}, v8f{}, v8f{}};

    #pragma unroll
    for (int kb = 0; kb < 2; ++kb) {
        // ---- Build B = 32x16 data tile once per kb (contiguous k order, k = 16*grp + e), reused
        //      across all 4 output row-tiles. Lanes 0-15: K=0..15 (V0-7),
        //      lanes 16-31: K=16..31; N = lane&15. f16 hi/lo split.
        v16h bh, bl;
        #pragma unroll
        for (int e = 0; e < 16; ++e) {
            int s = kb * 32 + grp * 16 + e;
            float xv = xs[s][jw * 16 + ln15];
            _Float16 hi = (_Float16)xv;
            bh[e] = hi;
            bl[e] = (_Float16)(xv - (float)hi);    // exact residual in f32
        }

        #pragma unroll
        for (int i = 0; i < 4; ++i) {
            if (kb * 32 + 32 <= i * 16) continue;  // statically pruned zero tiles

            // ---- A = 16x32 mask tile (same contiguous k order as B), synthesized in regs.
            // lanes 0-15: M=lane, K=0..15
            // lanes 16-31: M=lane-16, K=16..31
            int trow = i * 16 + ln15;
            v16h a;
            #pragma unroll
            for (int e = 0; e < 16; ++e) {
                int kl = 16 * grp + e;
                int s  = kb * 32 + kl;
                a[e] = (s >= trow) ? (_Float16)0.5f : (_Float16)0.0f;
            }

            acc[i] = __builtin_amdgcn_wmma_f32_16x16x32_f16(
                         false, a, false, bh, (short)0, acc[i], false, false);
            acc[i] = __builtin_amdgcn_wmma_f32_16x16x32_f16(
                         false, a, false, bl, (short)0, acc[i], false, false);
        }
    }

    // ---- Store C tiles (ISA C/D layout: VGPR r -> row 16i+8*grp+r, N=lane&15)
    float* op = out + base + jw * 16 + ln15;
    #pragma unroll
    for (int i = 0; i < 4; ++i) {
        #pragma unroll
        for (int r = 0; r < 8; ++r) {
            int t = i * 16 + grp * 8 + r;
            op[t * NH] = acc[i][r];
        }
    }
}

extern "C" void kernel_launch(void* const* d_in, const int* in_sizes, int n_in,
                              void* d_out, int out_size, void* d_ws, size_t ws_size,
                              hipStream_t stream) {
    const float* g   = (const float*)d_in[0];
    float*       out = (float*)d_out;
    int nchunks = in_sizes[0] / (BT * NH);   // (B*T*H)/(64*64) = 4096
    chunk_rev_cumsum_wmma<<<nchunks, 128, 0, stream>>>(g, out);
}
